// Decoder_command_27358941675644
// MI455X (gfx1250) — matrix operand order repaired and checked by reading, not compiled, not run on hardware
//
#include <hip/hip_runtime.h>
#include <cstddef>
#include <cstdint>
#include <math.h>

// ---------------- CDNA5 (gfx1250) types ----------------
typedef __attribute__((ext_vector_type(16))) __bf16 v16bf;
typedef __attribute__((ext_vector_type(8)))  __bf16 v8bf;
typedef __attribute__((ext_vector_type(8)))  float  v8f;

union ABf { v16bf v; v8bf h[2]; };

#define B_      64
#define S_SRC_  30
#define T_VID_  40
#define H_      1024
#define E_      300
#define V_      12000
#define MAXLEN_ 24
#define XCOLS_  2348            // emb(300) + vid_ctx(1024) + src_ctx(1024)
#define KL0_    3392            // XCOLS_ + H_ (=3372) padded to multiple of 32
#define KL1_    2048            // [h0_new | h1_old]
#define G_      4096            // 4*H gates

// ---------------- weight prep ----------------
__global__ void k_cast(const float* __restrict__ s, __bf16* __restrict__ d, int n) {
  int i = blockIdx.x * 256 + threadIdx.x;
  if (i < n) d[i] = (__bf16)s[i];
}

__global__ void k_pack_l0(const float* __restrict__ wih, const float* __restrict__ whh,
                          __bf16* __restrict__ d) {
  int i = blockIdx.x * 256 + threadIdx.x;
  if (i >= G_ * KL0_) return;
  int j = i / KL0_, k = i % KL0_;
  float v = 0.f;
  if (k < XCOLS_)           v = wih[(size_t)j * XCOLS_ + k];
  else if (k < XCOLS_ + H_) v = whh[(size_t)j * H_ + (k - XCOLS_)];
  d[i] = (__bf16)v;   // pad cols stay exact zero -> padded A cols are don't-care
}

__global__ void k_pack_l1(const float* __restrict__ wih, const float* __restrict__ whh,
                          __bf16* __restrict__ d) {
  int i = blockIdx.x * 256 + threadIdx.x;
  if (i >= G_ * KL1_) return;
  int j = i / KL1_, k = i % KL1_;
  float v = (k < H_) ? wih[(size_t)j * H_ + k] : whh[(size_t)j * H_ + (k - H_)];
  d[i] = (__bf16)v;
}

__global__ void k_addvec(const float* __restrict__ a, const float* __restrict__ b,
                         float* __restrict__ o, int n) {
  int i = blockIdx.x * 256 + threadIdx.x;
  if (i < n) o[i] = a[i] + b[i];
}

__global__ void k_init(const float* __restrict__ h0in, const float* __restrict__ c0in,
                       __bf16* __restrict__ h0bf, __bf16* __restrict__ h1bf,
                       float* __restrict__ c0s, float* __restrict__ c1s,
                       int* __restrict__ tok) {
  int i = blockIdx.x * 256 + threadIdx.x;
  if (i < B_ * H_) {
    h0bf[i] = (__bf16)h0in[i];
    h1bf[i] = (__bf16)h0in[B_ * H_ + i];
    c0s[i]  = c0in[i];
    c1s[i]  = c0in[B_ * H_ + i];
  }
  if (i < B_) tok[i] = 2;  // SOS
}

__global__ void k_zero_out0(float* __restrict__ out) {
  int i = blockIdx.x * 256 + threadIdx.x;
  if (i >= B_ * V_) return;
  int b = i / V_, v = i % V_;
  out[(size_t)b * (MAXLEN_ * V_) + v] = 0.f;
}

// ---------------- per-step small kernels ----------------
__global__ void k_build_x(const int* __restrict__ tok, const float* __restrict__ embed,
                          const __bf16* __restrict__ h0bf, const __bf16* __restrict__ h1bf,
                          __bf16* __restrict__ xh, __bf16* __restrict__ xh1) {
  int i = blockIdx.x * 256 + threadIdx.x;
  if (i >= B_ * XCOLS_) return;
  int b = i / XCOLS_, j = i % XCOLS_;
  if (j < E_) {
    xh[(size_t)b * KL0_ + j] = (__bf16)embed[(size_t)tok[b] * E_ + j];
  } else if (j < E_ + H_) {
    int c = j - E_;
    xh[(size_t)b * KL0_ + XCOLS_ + c] = h0bf[b * H_ + c];     // old h0 tail of x
  } else {
    int c = j - (E_ + H_);
    xh1[(size_t)b * KL1_ + H_ + c] = h1bf[b * H_ + c];        // old h1 for layer 1
  }
}

__global__ void k_score(const float* __restrict__ target,  // (64,1024)
                        const float* __restrict__ ctx,     // (64,S,1024)
                        const int* __restrict__ masktok,   // (64,S) src tokens or null
                        int S, float* __restrict__ scores) {
  int gid  = blockIdx.x * 8 + (threadIdx.x >> 5);
  int lane = threadIdx.x & 31;
  if (gid >= B_ * S) return;
  int b = gid / S, s = gid % S;
  const float* t = target + (size_t)b * H_;
  const float* c = ctx + ((size_t)b * S + s) * H_;
  float sum = 0.f;
  for (int h = lane; h < H_; h += 32) sum += t[h] * c[h];
#pragma unroll
  for (int off = 16; off > 0; off >>= 1) sum += __shfl_xor(sum, off, 32);
  if (lane == 0) {
    if (masktok && masktok[b * S + s] == 0) sum = -INFINITY;  // PAD mask
    scores[b * S + s] = sum;
  }
}

__global__ void k_softmax_ctx(const float* __restrict__ scores,
                              const float* __restrict__ ctx, // (64,S,1024)
                              __bf16* __restrict__ xh, int S, int colBase) {
  __shared__ float sc[64];
  __shared__ float w[64];
  int b = blockIdx.x, tid = threadIdx.x;
  if (tid < S) sc[tid] = scores[b * S + tid];
  __syncthreads();
  float m = -INFINITY;
  for (int s = 0; s < S; ++s) m = fmaxf(m, sc[s]);
  if (tid < S) w[tid] = expf(sc[tid] - m);
  __syncthreads();
  float den = 0.f;
  for (int s = 0; s < S; ++s) den += w[s];
  float inv = 1.f / den;
  for (int h = tid; h < H_; h += 256) {
    const float* c = ctx + (size_t)b * S * H_ + h;
    float acc = 0.f;
    for (int s = 0; s < S; ++s) acc += w[s] * c[(size_t)s * H_];
    xh[(size_t)b * KL0_ + colBase + h] = (__bf16)(acc * inv);
  }
}

__global__ void k_lstm_gate(const float* __restrict__ g, float* __restrict__ c,
                            __bf16* __restrict__ hbf, __bf16* __restrict__ hbf2, int ld2) {
  int i = blockIdx.x * 256 + threadIdx.x;
  if (i >= B_ * H_) return;
  int b = i >> 10, j = i & (H_ - 1);
  const float* gb = g + (size_t)b * G_;
  float ig = 1.f / (1.f + expf(-gb[j]));
  float fg = 1.f / (1.f + expf(-gb[H_ + j]));
  float gg = tanhf(gb[2 * H_ + j]);
  float og = 1.f / (1.f + expf(-gb[3 * H_ + j]));
  float cn = fg * c[i] + ig * gg;
  c[i] = cn;
  float hn = og * tanhf(cn);
  __bf16 hb = (__bf16)hn;
  hbf[i] = hb;
  if (hbf2) hbf2[(size_t)b * ld2 + j] = hb;
}

__global__ void k_argmax(const float* __restrict__ logits, int ld, int* __restrict__ tok) {
  __shared__ float bv[256];
  __shared__ int   bi[256];
  int b = blockIdx.x, tid = threadIdx.x;
  const float* row = logits + (size_t)b * ld;
  float best = -INFINITY; int bidx = 0;
  for (int v = tid; v < V_; v += 256) {
    float x = row[v];
    if (x > best) { best = x; bidx = v; }   // strictly greater -> first max kept
  }
  bv[tid] = best; bi[tid] = bidx;
  __syncthreads();
  for (int off = 128; off > 0; off >>= 1) {
    if (tid < off) {
      if (bv[tid + off] > bv[tid] ||
          (bv[tid + off] == bv[tid] && bi[tid + off] < bi[tid])) {
        bv[tid] = bv[tid + off]; bi[tid] = bi[tid + off];
      }
    }
    __syncthreads();
  }
  if (tid == 0) tok[b] = bi[0];
}

// ---------------- bf16 WMMA GEMM:  Y(64,N) = X(64,K) @ W(N,K)^T + bias ----------------
// block = 4 waves; A K-chunk (64x32 bf16 = 4KB) double-buffered in LDS and shared by
// 4 waves x 4 N-tiles = 16 output tiles, so B (weight) traffic dominates as it should.
// tileBase is forced into an SGPR so the tile guards are scalar branches and EXEC
// stays all-ones around v_wmma (ISA requirement), with one barrier per K-chunk.
__global__ __launch_bounds__(128)
void k_gemm(const __bf16* __restrict__ X, const __bf16* __restrict__ W,
            const float* __restrict__ bias, float* __restrict__ Y,
            int ldY, int N, int K) {
  __shared__ alignas(32) __bf16 As[2][64][32];
  const int tid  = threadIdx.x;
  const int lane = tid & 31;
  const int lh   = lane & 15;   // row-in-tile (A) / column (B,C,D)
  const int kh   = lane >> 4;   // half-wave selector
  const int tileBase =
      __builtin_amdgcn_readfirstlane((blockIdx.x * 4 + (tid >> 5)) * 4);
  const int Ntiles = N >> 4;

  const v8f vzero = {0.f, 0.f, 0.f, 0.f, 0.f, 0.f, 0.f, 0.f};
  v8f acc[4][4];
#pragma unroll
  for (int nt = 0; nt < 4; ++nt)
#pragma unroll
    for (int mt = 0; mt < 4; ++mt) acc[nt][mt] = vzero;

  const int rowA = tid >> 1;          // 0..63
  const int segA = (tid & 1) * 16;    // 0 or 16

  // software-pipelined A staging (global -> reg -> LDS), double buffered
  v16bf stage = *(const v16bf*)&X[(size_t)rowA * K + segA];
  int p = 0;

  for (int k0 = 0; k0 < K; k0 += 32) {
    *(v16bf*)&As[p][rowA][segA] = stage;
    __syncthreads();
    if (k0 + 32 < K)
      stage = *(const v16bf*)&X[(size_t)rowA * K + (k0 + 32) + segA];

    // A fragments in contiguous K order (lane half kh holds K = kh*16 .. +15), same as B
    ABf a[4];
#pragma unroll
    for (int mt = 0; mt < 4; ++mt) {
      const __bf16* ap = &As[p][mt * 16 + lh][kh * 16];
      a[mt].h[0] = *(const v8bf*)ap;          // K = kh*16 .. +7
      a[mt].h[1] = *(const v8bf*)(ap + 8);    // K = kh*16+8 .. +15
    }

#pragma unroll
    for (int nt = 0; nt < 4; ++nt) {
      if (tileBase + nt < Ntiles) {           // scalar (SGPR) guard: s_cbranch only
        const int tile = tileBase + nt;
        const __bf16* wp = W + (size_t)(tile * 16 + lh) * K + k0 + kh * 16;
        if (k0 + 32 < K) __builtin_prefetch(wp + 32, 0, 1);
        ABf bf; bf.v = *(const v16bf*)wp;     // B 32x16: lane=N col, 16 contiguous K
#pragma unroll
        for (int mt = 0; mt < 4; ++mt)
          acc[nt][mt] = __builtin_amdgcn_wmma_f32_16x16x32_bf16(
              false, a[mt].v, false, bf.v, (short)0, acc[nt][mt], false, false);
      }
    }
    p ^= 1;
  }

  const int rbase = kh * 8;                   // C/D layout: upper half-wave = M+8
#pragma unroll
  for (int nt = 0; nt < 4; ++nt) {
    if (tileBase + nt >= Ntiles) continue;
    const int tile = tileBase + nt;
    const int gn = tile * 16 + lh;
    const float bvl = bias ? bias[gn] : 0.f;
#pragma unroll
    for (int mt = 0; mt < 4; ++mt)
#pragma unroll
      for (int d = 0; d < 8; ++d) {
        const int m = mt * 16 + rbase + d;
        Y[(size_t)m * ldY + gn] = acc[nt][mt][d] + bvl;
      }
  }
}

// ---------------- host orchestration ----------------
extern "C" void kernel_launch(void* const* d_in, const int* in_sizes, int n_in,
                              void* d_out_v, int out_size, void* d_ws, size_t ws_size,
                              hipStream_t stream) {
  (void)in_sizes; (void)n_in; (void)out_size; (void)ws_size;

  const int*   src   = (const int*)  d_in[0];
  const float* h0in  = (const float*)d_in[2];
  const float* c0in  = (const float*)d_in[3];
  const float* vctx  = (const float*)d_in[4];   // vid_out_rnn (64,40,1024)
  const float* sctx  = (const float*)d_in[5];   // src_out (64,30,1024)
  const float* emb   = (const float*)d_in[6];
  const float* Wsrc  = (const float*)d_in[7];
  const float* Wvid  = (const float*)d_in[8];
  const float* Wih0  = (const float*)d_in[9];
  const float* Whh0  = (const float*)d_in[10];
  const float* bih0  = (const float*)d_in[11];
  const float* bhh0  = (const float*)d_in[12];
  const float* Wih1  = (const float*)d_in[13];
  const float* Whh1  = (const float*)d_in[14];
  const float* bih1  = (const float*)d_in[15];
  const float* bhh1  = (const float*)d_in[16];
  const float* fcW   = (const float*)d_in[17];
  const float* fcb   = (const float*)d_in[18];
  float* out = (float*)d_out_v;

  char* ws = (char*)d_ws;
  size_t off = 0;
  auto alloc = [&](size_t bytes) -> void* {
    off = (off + 255) & ~(size_t)255;
    void* p = ws + off;
    off += bytes;
    return p;
  };

  __bf16* Wvid_bf = (__bf16*)alloc((size_t)H_ * H_ * 2);
  __bf16* Wsrc_bf = (__bf16*)alloc((size_t)H_ * H_ * 2);
  __bf16* Wl0_bf  = (__bf16*)alloc((size_t)G_ * KL0_ * 2);
  __bf16* Wl1_bf  = (__bf16*)alloc((size_t)G_ * KL1_ * 2);
  __bf16* fcW_bf  = (__bf16*)alloc((size_t)V_ * H_ * 2);
  float*  bias0   = (float*) alloc((size_t)G_ * 4);
  float*  bias1   = (float*) alloc((size_t)G_ * 4);
  __bf16* h0bf    = (__bf16*)alloc((size_t)B_ * H_ * 2);
  __bf16* h1bf    = (__bf16*)alloc((size_t)B_ * H_ * 2);
  float*  c0s     = (float*) alloc((size_t)B_ * H_ * 4);
  float*  c1s     = (float*) alloc((size_t)B_ * H_ * 4);
  __bf16* xh      = (__bf16*)alloc((size_t)B_ * KL0_ * 2);
  __bf16* xh1     = (__bf16*)alloc((size_t)B_ * KL1_ * 2);
  float*  tvid    = (float*) alloc((size_t)B_ * H_ * 4);
  float*  tsrc    = (float*) alloc((size_t)B_ * H_ * 4);
  float*  sc_v    = (float*) alloc((size_t)B_ * T_VID_ * 4);
  float*  sc_s    = (float*) alloc((size_t)B_ * S_SRC_ * 4);
  float*  g0      = (float*) alloc((size_t)B_ * G_ * 4);
  float*  g1      = (float*) alloc((size_t)B_ * G_ * 4);
  int*    tok     = (int*)   alloc((size_t)B_ * 4);

  // ---- one-time weight prep (per call; amortized over 23 steps) ----
  k_cast<<<(H_ * H_ + 255) / 256, 256, 0, stream>>>(Wvid, Wvid_bf, H_ * H_);
  k_cast<<<(H_ * H_ + 255) / 256, 256, 0, stream>>>(Wsrc, Wsrc_bf, H_ * H_);
  k_cast<<<(V_ * H_ + 255) / 256, 256, 0, stream>>>(fcW, fcW_bf, V_ * H_);
  k_pack_l0<<<(G_ * KL0_ + 255) / 256, 256, 0, stream>>>(Wih0, Whh0, Wl0_bf);
  k_pack_l1<<<(G_ * KL1_ + 255) / 256, 256, 0, stream>>>(Wih1, Whh1, Wl1_bf);
  k_addvec<<<(G_ + 255) / 256, 256, 0, stream>>>(bih0, bhh0, bias0, G_);
  k_addvec<<<(G_ + 255) / 256, 256, 0, stream>>>(bih1, bhh1, bias1, G_);
  k_init<<<(B_ * H_ + 255) / 256, 256, 0, stream>>>(h0in, c0in, h0bf, h1bf, c0s, c1s, tok);
  k_zero_out0<<<(B_ * V_ + 255) / 256, 256, 0, stream>>>(out);

  const int ldOut = MAXLEN_ * V_;

  for (int t = 0; t < MAXLEN_ - 1; ++t) {
    // x = [embed[tok] | vid_ctx | src_ctx | h0]; xh1 tail = old h1
    k_build_x<<<(B_ * XCOLS_ + 255) / 256, 256, 0, stream>>>(tok, emb, h0bf, h1bf, xh, xh1);

    // attention targets: h0 @ W^T  (WMMA)
    k_gemm<<<(H_ / 16 + 15) / 16, 128, 0, stream>>>(h0bf, Wvid_bf, nullptr, tvid, H_, H_, H_);
    k_gemm<<<(H_ / 16 + 15) / 16, 128, 0, stream>>>(h0bf, Wsrc_bf, nullptr, tsrc, H_, H_, H_);

    // scores + softmax-weighted contexts (written as bf16 straight into x)
    k_score<<<(B_ * T_VID_ + 7) / 8, 256, 0, stream>>>(tvid, vctx, nullptr, T_VID_, sc_v);
    k_score<<<(B_ * S_SRC_ + 7) / 8, 256, 0, stream>>>(tsrc, sctx, src, S_SRC_, sc_s);
    k_softmax_ctx<<<B_, 256, 0, stream>>>(sc_v, vctx, xh, T_VID_, E_);
    k_softmax_ctx<<<B_, 256, 0, stream>>>(sc_s, sctx, xh, S_SRC_, E_ + H_);

    // LSTM layer 0 (fused [x|h0] GEMM), gates
    k_gemm<<<(G_ / 16 + 15) / 16, 128, 0, stream>>>(xh, Wl0_bf, bias0, g0, G_, G_, KL0_);
    k_lstm_gate<<<(B_ * H_ + 255) / 256, 256, 0, stream>>>(g0, c0s, h0bf, xh1, KL1_);

    // LSTM layer 1 (fused [h0n|h1] GEMM), gates
    k_gemm<<<(G_ / 16 + 15) / 16, 128, 0, stream>>>(xh1, Wl1_bf, bias1, g1, G_, G_, KL1_);
    k_lstm_gate<<<(B_ * H_ + 255) / 256, 256, 0, stream>>>(g1, c1s, h1bf, nullptr, 0);

    // FC logits straight into d_out[:, t+1, :]
    float* lbase = out + (size_t)(t + 1) * V_;
    k_gemm<<<(V_ / 16 + 15) / 16, 128, 0, stream>>>(h1bf, fcW_bf, fcb, lbase, ldOut, V_, H_);

    // greedy token for next step
    if (t < MAXLEN_ - 2)
      k_argmax<<<B_, 256, 0, stream>>>(lbase, ldOut, tok);
  }
}
